// RNNEmbedding_11802570129992
// MI455X (gfx1250) — matrix operand order repaired and checked by reading, not compiled, not run on hardware
//
#include <hip/hip_runtime.h>
#include <hip/hip_bf16.h>

typedef __bf16 bf16_t;
typedef __attribute__((ext_vector_type(16))) __bf16 v16bf;
typedef __attribute__((ext_vector_type(8)))  float  v8f;
typedef __attribute__((ext_vector_type(4)))  float  f32x4;

#define T_STEPS 512
#define BATCH   512
#define DDIM    300
#define HDIM    128
#define EGO_    12
#define GDIM    512          // 4*H
#define KA      160          // padded LSTM K: 128 (h) + 12 (x) -> 160
#define BT      16           // batch rows per workgroup
#define GSTRIDE (GDIM + 4)   // padded gates stride (bank-conflict relief)
#define KS      320          // padded state-feature K: 300 -> 320
#define KF      448          // total padded K: 320 (state) + 128 (hs)
#define EDIM    64

union FragB16 { unsigned int u[8]; v16bf v; };
union FragF32 { float f[8]; v8f v; };
union Pack4   { bf16_t h[4]; unsigned long long u; };

__device__ __forceinline__ unsigned int pack2bf(float a, float b) {
  union { bf16_t h[2]; unsigned int u; } p;
  p.h[0] = (bf16_t)a; p.h[1] = (bf16_t)b;
  return p.u;
}

__device__ __forceinline__ v8f zero_v8f() {
  FragF32 z;
#pragma unroll
  for (int r = 0; r < 8; ++r) z.f[r] = 0.f;
  return z.v;
}

__device__ __forceinline__ float fast_sigmoid(float x) {
  return __builtin_amdgcn_rcpf(1.f + __expf(-x));
}
__device__ __forceinline__ float fast_tanh(float x) {
#if __has_builtin(__builtin_amdgcn_tanhf)
  return __builtin_amdgcn_tanhf(x);          // native V_TANH_F32
#else
  return 2.f * fast_sigmoid(2.f * x) - 1.f;  // branchless fallback
#endif
}

// bf16 A-matrix 16x32 fragment (contiguous K order, not the ISA table order): lane l -> M = l&15, K-half = l>>4.
__device__ __forceinline__ v16bf load_A_frag(const bf16_t* base, int stride,
                                             int row0, int k0, int lane) {
  const int half = lane >> 4;
  const int m    = lane & 15;
  FragB16 fr;
#pragma unroll
  for (int v = 0; v < 8; ++v) {
    const int kb = half * 16 + 2 * v;
    fr.u[v] = *(const unsigned int*)(base + (row0 + m) * stride + k0 + kb);
  }
  return fr.v;
}

// bf16 B-matrix 32x16 fragment: lane l -> N = l&15, K range = (l>>4)*16 + 2v.
__device__ __forceinline__ v16bf load_B_frag(const bf16_t* base, int stride,
                                             int k0, int n0, int lane) {
  const int half = lane >> 4;
  const int n    = lane & 15;
  FragB16 fr;
#pragma unroll
  for (int v = 0; v < 8; ++v) {
    const int kb = half * 16 + 2 * v;
    fr.u[v] = *(const unsigned int*)(base + (n0 + n) * stride + k0 + kb);
  }
  return fr.v;
}

// ---------------------------------------------------------------- K0: offsets
__global__ void scan_kernel(const int* __restrict__ lens, int* __restrict__ offs) {
  if (threadIdx.x == 0 && blockIdx.x == 0) {
    int acc = 0;
    for (int i = 0; i < T_STEPS; ++i) { offs[i] = acc; acc += lens[i]; }
    offs[T_STEPS] = acc;
  }
}

// ---------------------------------------------------------------- K1: LSTM
// grid = B/BT = 32 workgroups, block = 512 threads (16 waves).
// Wave w owns gate columns [32w, 32w+32); weights live in registers (10 B-frags).
// 4 independent WMMA chains (2 N-tiles x 2 K-parity chains) to avoid hazard NOPs.
__global__ __launch_bounds__(512)
void lstm_kernel(const float* __restrict__ state,
                 const float* __restrict__ h0,
                 const float* __restrict__ c0,
                 const float* __restrict__ W_ih,
                 const float* __restrict__ W_hh,
                 const float* __restrict__ b_ih,
                 const float* __restrict__ b_hh,
                 bf16_t* __restrict__ hs)        // [T][B][H] bf16
{
  __shared__ __align__(16) bf16_t Ah[BT][KA];    // [h(128) | x_t(12) | zeros]
  __shared__ float Gt[BT][GSTRIDE];              // gate accumulators (f32)

  const int tid  = threadIdx.x;
  const int lane = tid & 31;
  const int wave = tid >> 5;                     // 0..15
  const int b0   = blockIdx.x * BT;

  // ---- weight-stationary B fragments: 2 N-tiles x 5 K-tiles (K padded to 160)
  v16bf Bfrag[10];
  {
    const int half = lane >> 4;
    const int n_l  = lane & 15;
#pragma unroll
    for (int nt = 0; nt < 2; ++nt) {
      const int n = wave * 32 + nt * 16 + n_l;   // gate row in W (0..511)
#pragma unroll
      for (int kt = 0; kt < 5; ++kt) {
        FragB16 fr;
#pragma unroll
        for (int v = 0; v < 8; ++v) {
          const int k = kt * 32 + half * 16 + 2 * v;   // K in [0,160)
          float x0 = 0.f, x1 = 0.f;
          if (k < HDIM) {                              // recurrent weights
            x0 = W_hh[n * HDIM + k];
            x1 = W_hh[n * HDIM + k + 1];
          } else {                                     // input weights (K 128..139)
            const int kk = k - HDIM;
            if (kk < EGO_) {
              x0 = W_ih[n * EGO_ + kk];
              if (kk + 1 < EGO_) x1 = W_ih[n * EGO_ + kk + 1];
            }
          }
          fr.u[v] = pack2bf(x0, x1);
        }
        Bfrag[nt * 5 + kt] = fr.v;
      }
    }
  }

  // ---- per-thread cell state + fused biases (4 consecutive j per thread)
  const int flat0 = tid * 4;                     // 0..2044, same batch row for e=0..3
  const int brow  = flat0 >> 7;
  const int j0    = flat0 & 127;
  float c_reg[4];
  float bsum[4][4];
#pragma unroll
  for (int e = 0; e < 4; ++e) {
    const int j = j0 + e;
    c_reg[e] = c0[(size_t)(b0 + brow) * HDIM + j];
#pragma unroll
    for (int g = 0; g < 4; ++g)
      bsum[e][g] = b_ih[g * HDIM + j] + b_hh[g * HDIM + j];
    Ah[brow][j] = (bf16_t)h0[(size_t)(b0 + brow) * HDIM + j];
  }
  if (tid < BT * 20) {                           // zero-pad K 140..159 (once)
    Ah[tid / 20][140 + tid % 20] = (bf16_t)0.f;
  }
  if (tid < BT * EGO_) {                         // x_0
    const int b = tid / EGO_, d = tid % EGO_;
    Ah[b][HDIM + d] = (bf16_t)state[(size_t)(b0 + b) * DDIM + d];
  }
  __syncthreads();

  for (int t = 0; t < T_STEPS; ++t) {
    // A fragments for [h | x | pad]
    v16bf Afr[5];
#pragma unroll
    for (int kt = 0; kt < 5; ++kt)
      Afr[kt] = load_A_frag(&Ah[0][0], KA, 0, kt * 32, lane);

    // 4 independent accumulation chains: [nt][kt&1]
    v8f acc[2][2];
#pragma unroll
    for (int nt = 0; nt < 2; ++nt)
#pragma unroll
      for (int ch = 0; ch < 2; ++ch) acc[nt][ch] = zero_v8f();

#pragma unroll
    for (int kt = 0; kt < 5; ++kt) {
      const int ch = kt & 1;
#pragma unroll
      for (int nt = 0; nt < 2; ++nt)
        acc[nt][ch] = __builtin_amdgcn_wmma_f32_16x16x32_bf16(
            false, Afr[kt], false, Bfrag[nt * 5 + kt], (short)0, acc[nt][ch],
            false, false);
    }

#pragma unroll
    for (int nt = 0; nt < 2; ++nt) {
      FragF32 r0, r1;
      r0.v = acc[nt][0]; r1.v = acc[nt][1];
      const int n_l  = lane & 15;
      const int half = lane >> 4;
      const int ng   = wave * 32 + nt * 16 + n_l;
#pragma unroll
      for (int r = 0; r < 8; ++r)
        Gt[r + half * 8][ng] = r0.f[r] + r1.f[r];  // C layout: M = r + 8*half
    }
    __syncthreads();                              // gates complete; A reads done

    // elementwise LSTM cell (native tanh / rcp+exp sigmoid, all straight-line)
    Pack4 hp;
#pragma unroll
    for (int e = 0; e < 4; ++e) {
      const int j = j0 + e;
      const float gi = Gt[brow][j]            + bsum[e][0];
      const float gf = Gt[brow][HDIM + j]     + bsum[e][1];
      const float gg = Gt[brow][2 * HDIM + j] + bsum[e][2];
      const float go = Gt[brow][3 * HDIM + j] + bsum[e][3];
      const float cn = fast_sigmoid(gf) * c_reg[e] + fast_sigmoid(gi) * fast_tanh(gg);
      c_reg[e] = cn;
      hp.h[e] = (bf16_t)(fast_sigmoid(go) * fast_tanh(cn));
    }
    *(unsigned long long*)&Ah[brow][j0] = hp.u;                 // 8B LDS store
    *(unsigned long long*)(hs + ((size_t)t * BATCH + (b0 + brow)) * HDIM + j0) = hp.u;

    if (t + 1 < T_STEPS && tid < BT * EGO_) {     // x_{t+1}
      const int b = tid / EGO_, d = tid % EGO_;
      Ah[b][HDIM + d] =
          (bf16_t)state[((size_t)(t + 1) * BATCH + (b0 + b)) * DDIM + d];
    }
    if (t + 2 < T_STEPS && tid < BT) {            // prefetch x_{t+2} row
      __builtin_prefetch(&state[((size_t)(t + 2) * BATCH + (b0 + tid)) * DDIM], 0, 3);
    }
    __syncthreads();
  }
}

// ---------------------------------------------------------------- K2: pack+GEMM
// grid = 512 groups, block = 256 threads (8 waves = 2 M-tiles x 4 N-tiles).
// K layout: [0..319] = state features (300 + pad), [320..447] = hs features.
__global__ __launch_bounds__(256)
void emb_kernel(const float* __restrict__ state,
                const bf16_t* __restrict__ hs,
                const float* __restrict__ W_emb,
                const float* __restrict__ b_emb,
                const int* __restrict__ lens,
                const int* __restrict__ offs,
                float* __restrict__ out)
{
  __shared__ __align__(16) bf16_t Wb[EDIM][KF];   // 56 KB (W_emb, bf16, remapped K)
  __shared__ __align__(16) bf16_t A2s[32][KS];    // 20 KB (state part of tile)
  __shared__ __align__(16) bf16_t A2h[32][HDIM];  //  8 KB (hs part, async-copied)
  __shared__ float be[EDIM];

  const int a    = blockIdx.x;
  const int tid  = threadIdx.x;
  const int lane = tid & 31;
  const int wave = tid >> 5;
  const int mt   = wave >> 2;                     // 0..1
  const int nt   = wave & 3;                      // 0..3
  const int L    = lens[a];
  const int base = offs[a];

  for (int idx = tid; idx < EDIM * KF; idx += 256) {
    const int n = idx / KF, k = idx - n * KF;
    float w = 0.f;
    if (k < DDIM)      w = W_emb[(size_t)n * (DDIM + HDIM) + k];
    else if (k >= KS)  w = W_emb[(size_t)n * (DDIM + HDIM) + DDIM + (k - KS)];
    Wb[n][k] = (bf16_t)w;
  }
  if (tid < EDIM) be[tid] = b_emb[tid];
  __syncthreads();

  for (int c0 = 0; c0 < L; c0 += 32) {
    // hs tile: rows c0..c0+31 are one contiguous bf16 block -> async copy to LDS
    {
      const char* src = (const char*)(hs + ((size_t)a * BATCH + c0) * HDIM);
      const unsigned ldsbase = (unsigned)(size_t)&A2h[0][0];
#pragma unroll
      for (int it = 0; it < 2; ++it) {
        const unsigned off = (unsigned)(tid + it * 256) * 16u;   // 2 x 16B per thread
        asm volatile("global_load_async_to_lds_b128 %0, %1, off"
                     :: "v"(ldsbase + off), "v"(src + off) : "memory");
      }
    }
    // state tile: 8 threads per row, float4 loads -> packed bf16 8B stores
    {
      const int r = tid >> 3;
      const int c = c0 + r;
      const int ksub = (tid & 7) * 40;
      const float* srow = state + ((size_t)c * BATCH + a) * DDIM;
      const bool valid = (c < L);
#pragma unroll
      for (int i = 0; i < 40; i += 4) {
        const int k = ksub + i;
        Pack4 p;
        if (valid && k < DDIM) {
          const f32x4 v4 = *(const f32x4*)(srow + k);
          p.h[0] = (bf16_t)v4.x; p.h[1] = (bf16_t)v4.y;
          p.h[2] = (bf16_t)v4.z; p.h[3] = (bf16_t)v4.w;
        } else {
          p.u = 0ull;
        }
        *(unsigned long long*)&A2s[r][k] = p.u;
      }
    }
    asm volatile("s_wait_asynccnt 0" ::: "memory");
    __syncthreads();

    // 2 independent accumulation chains over 14 K-tiles
    v8f accA = zero_v8f();
    v8f accB = zero_v8f();
#pragma unroll
    for (int kt = 0; kt < 10; ++kt) {             // state K-tiles
      const v16bf Af = load_A_frag(&A2s[0][0], KS, mt * 16, kt * 32, lane);
      const v16bf Bf = load_B_frag(&Wb[0][0], KF, kt * 32, nt * 16, lane);
      if (kt & 1)
        accB = __builtin_amdgcn_wmma_f32_16x16x32_bf16(
            false, Af, false, Bf, (short)0, accB, false, false);
      else
        accA = __builtin_amdgcn_wmma_f32_16x16x32_bf16(
            false, Af, false, Bf, (short)0, accA, false, false);
    }
#pragma unroll
    for (int kt = 0; kt < 4; ++kt) {              // hs K-tiles
      const v16bf Af = load_A_frag(&A2h[0][0], HDIM, mt * 16, kt * 32, lane);
      const v16bf Bf = load_B_frag(&Wb[0][0], KF, KS + kt * 32, nt * 16, lane);
      if (kt & 1)
        accB = __builtin_amdgcn_wmma_f32_16x16x32_bf16(
            false, Af, false, Bf, (short)0, accB, false, false);
      else
        accA = __builtin_amdgcn_wmma_f32_16x16x32_bf16(
            false, Af, false, Bf, (short)0, accA, false, false);
    }

    FragF32 r0, r1;
    r0.v = accA; r1.v = accB;
    const int n_l  = lane & 15;
    const int half = lane >> 4;
    const int n    = nt * 16 + n_l;
    const float bias = be[n];
#pragma unroll
    for (int r = 0; r < 8; ++r) {
      const int c = c0 + mt * 16 + r + half * 8;
      if (c < L)
        out[(size_t)(base + c) * EDIM + n] = r0.f[r] + r1.f[r] + bias;
    }
    __syncthreads();
  }
}

// ---------------------------------------------------------------- launcher
extern "C" void kernel_launch(void* const* d_in, const int* in_sizes, int n_in,
                              void* d_out, int out_size, void* d_ws, size_t ws_size,
                              hipStream_t stream) {
  const float* state = (const float*)d_in[0];
  const float* h0    = (const float*)d_in[1];
  const float* c0    = (const float*)d_in[2];
  const int*   lens  = (const int*)d_in[3];
  const float* W_ih  = (const float*)d_in[4];
  const float* W_hh  = (const float*)d_in[5];
  const float* b_ih  = (const float*)d_in[6];
  const float* b_hh  = (const float*)d_in[7];
  const float* W_emb = (const float*)d_in[8];
  const float* b_emb = (const float*)d_in[9];
  float* out = (float*)d_out;

  // ws layout: [ hs bf16: T*B*H*2 = 64 MB ][ offsets: (T+1) ints ]
  bf16_t* hs = (bf16_t*)d_ws;
  int* offs = (int*)((char*)d_ws + (size_t)T_STEPS * BATCH * HDIM * sizeof(bf16_t));

  scan_kernel<<<dim3(1), dim3(64), 0, stream>>>(lens, offs);
  lstm_kernel<<<dim3(BATCH / BT), dim3(512), 0, stream>>>(
      state, h0, c0, W_ih, W_hh, b_ih, b_hh, hs);
  emb_kernel<<<dim3(T_STEPS), dim3(256), 0, stream>>>(
      state, hs, W_emb, b_emb, lens, offs, out);
}
